// SAGE_83502754168881
// MI455X (gfx1250) — matrix operand order repaired and checked by reading, not compiled, not run on hardware
//
#include <hip/hip_runtime.h>

#define N_NODES 50000
#define N_EDGES 1600000
#define HID 512
#define NEG_SLOPE 0.01f
// rows padded so every 256-row GEMM block tile is fully addressable
#define M_PAD 50176   // 196 * 256

typedef __bf16 bf16;
typedef __attribute__((ext_vector_type(16))) __bf16 v16bf;
typedef __attribute__((ext_vector_type(8)))  __bf16 v8bf;
typedef __attribute__((ext_vector_type(8)))  float  v8f;

// ---------------- elementwise helpers ----------------

__global__ void zero_f32(float* p, long long n) {
    long long t = (long long)blockIdx.x * blockDim.x + threadIdx.x;
    if (t < n) p[t] = 0.0f;
}

__global__ void f32_to_bf16(const float* __restrict__ in, bf16* __restrict__ out, long long n) {
    long long t = (long long)blockIdx.x * blockDim.x + threadIdx.x;
    if (t < n) out[t] = (bf16)in[t];
}

// Wt[n*512+k] = (bf16)W[k*512+n]  (transpose so each lane's B frag is contiguous)
__global__ void wt_transpose(const float* __restrict__ W, bf16* __restrict__ Wt) {
    int t = blockIdx.x * blockDim.x + threadIdx.x;   // 512*512 threads
    int n = t >> 9, k = t & 511;
    Wt[t] = (bf16)W[k * HID + n];
}

// ---------------- graph aggregation ----------------

__global__ void count_edges(const int* __restrict__ dst, float* __restrict__ cnt) {
    int e = blockIdx.x * blockDim.x + threadIdx.x;
    if (e < N_EDGES) unsafeAtomicAdd(&cnt[dst[e]], 1.0f);
}

// agg[dst] += X[src] ; 128 threads per edge, 4 columns each (f32 atomics, L2-resident)
__global__ void scatter_bf16(const bf16* __restrict__ X, const int* __restrict__ src,
                             const int* __restrict__ dst, float* __restrict__ agg) {
    long long t = (long long)blockIdx.x * blockDim.x + threadIdx.x;
    int e = (int)(t >> 7);
    if (e >= N_EDGES) return;
    int c = ((int)t & 127) << 2;
    int s = src[e], d = dst[e];
    const bf16* xp = X + (size_t)s * HID + c;
    float* ap = agg + (size_t)d * HID + c;
    float v0 = (float)xp[0], v1 = (float)xp[1], v2 = (float)xp[2], v3 = (float)xp[3];
    unsafeAtomicAdd(ap + 0, v0);
    unsafeAtomicAdd(ap + 1, v1);
    unsafeAtomicAdd(ap + 2, v2);
    unsafeAtomicAdd(ap + 3, v3);
}

__global__ void mean_bf16(const float* __restrict__ agg, const float* __restrict__ cnt,
                          bf16* __restrict__ out) {
    long long t = (long long)blockIdx.x * blockDim.x + threadIdx.x;
    long long n = (long long)N_NODES * HID;
    if (t >= n) return;
    float c = cnt[t >> 9];
    c = (c < 1.0f) ? 1.0f : c;
    out[t] = (bf16)(agg[t] / c);
}

// ---------------- bf16 WMMA GEMM ----------------
// out[M,512](bf16) = epi( A1@B1 (+ A2@B2) + bias ), Bt* are [N][K] transposed bf16.
// Wave tile: 32 rows x 128 cols (16 f32 accumulators).  Block = 8 waves = 256 rows.
// All 20 loads per K-step come off 2 base pointers with immediate offsets.
__global__ __launch_bounds__(256)
void gemm_bf16(const bf16* __restrict__ A1, const bf16* __restrict__ Bt1,
               const bf16* __restrict__ A2, const bf16* __restrict__ Bt2,
               const float* __restrict__ bias, bf16* __restrict__ out,
               int M, int lrelu) {
    const int lane = threadIdx.x & 31;
    const int wave = threadIdx.x >> 5;
    const int rowBase = (blockIdx.y * 8 + wave) * 32;   // < M_PAD by construction
    const int colBase = blockIdx.x * 128;
    const int h   = lane >> 4;
    const int l16 = lane & 15;
    const int n0  = colBase + l16;

    v8f acc0[8], acc1[8];
#pragma unroll
    for (int t = 0; t < 8; ++t)
#pragma unroll
        for (int i = 0; i < 8; ++i) { acc0[t][i] = 0.0f; acc1[t][i] = 0.0f; }

#pragma unroll 1
    for (int pass = 0; pass < 2; ++pass) {
        const bf16* A  = pass ? A2  : A1;
        const bf16* Bt = pass ? Bt2 : Bt1;
        if (A == nullptr) break;              // uniform
        // single base pointers; everything else is a compile-time byte offset
        const bf16* aptr = A  + (size_t)(rowBase + l16) * HID + 16 * h;
        const bf16* bptr = Bt + (size_t)n0 * HID + 16 * h;
#pragma unroll 1
        for (int k0 = 0; k0 < HID; k0 += 32) {
            // A frags (contiguous K order, matching B): K = k0+16h..+15
            v8bf a0lo = *(const v8bf*)(aptr);
            v8bf a0hi = *(const v8bf*)(aptr + 8);
            v8bf a1lo = *(const v8bf*)(aptr + 16 * HID);        // +16384 B
            v8bf a1hi = *(const v8bf*)(aptr + 16 * HID + 8);
            v16bf a0 = __builtin_shufflevector(a0lo, a0hi,
                0, 1, 2, 3, 4, 5, 6, 7, 8, 9, 10, 11, 12, 13, 14, 15);
            v16bf a1 = __builtin_shufflevector(a1lo, a1hi,
                0, 1, 2, 3, 4, 5, 6, 7, 8, 9, 10, 11, 12, 13, 14, 15);
#pragma unroll
            for (int t = 0; t < 8; ++t) {
                // B frag: column n0+16t, K = k0+16h..+15 ; tile stride 16384 B (imm)
                v16bf b = *(const v16bf*)(bptr + (size_t)t * 16 * HID);
                acc0[t] = __builtin_amdgcn_wmma_f32_16x16x32_bf16(
                    false, a0, false, b, (short)0, acc0[t], false, false);
                acc1[t] = __builtin_amdgcn_wmma_f32_16x16x32_bf16(
                    false, a1, false, b, (short)0, acc1[t], false, false);
            }
            aptr += 32;
            bptr += 32;
        }
    }

    // Epilogue: C element (g, lane) of row-tile r is row rowBase+16r+8h+g,
    // col colBase+16t+l16.  Store guard handles the ragged final tiles.
#pragma unroll
    for (int t = 0; t < 8; ++t) {
        int n = colBase + 16 * t + l16;
        float bv = bias[n];
#pragma unroll
        for (int g = 0; g < 8; ++g) {
            int mr0 = rowBase + 8 * h + g;
            float v0 = acc0[t][g] + bv;
            if (lrelu) v0 = (v0 >= 0.0f) ? v0 : NEG_SLOPE * v0;
            if (mr0 < M) out[(size_t)mr0 * HID + n] = (bf16)v0;
            int mr1 = mr0 + 16;
            float v1 = acc1[t][g] + bv;
            if (lrelu) v1 = (v1 >= 0.0f) ? v1 : NEG_SLOPE * v1;
            if (mr1 < M) out[(size_t)mr1 * HID + n] = (bf16)v1;
        }
    }
}

// ---------------- tiny output head: [50000,512] @ [512,4] + b ----------------
__global__ void out_head(const bf16* __restrict__ X, const float* __restrict__ Wo,
                         const float* __restrict__ bo, float* __restrict__ out) {
    int t = blockIdx.x * blockDim.x + threadIdx.x;
    if (t >= N_NODES * 4) return;
    int m = t >> 2, j = t & 3;
    const bf16* xr = X + (size_t)m * HID;
    float s = bo[j];
#pragma unroll 8
    for (int k = 0; k < HID; ++k) s += (float)xr[k] * Wo[(k << 2) + j];
    out[t] = s;
}

// ---------------- orchestration ----------------

extern "C" void kernel_launch(void* const* d_in, const int* in_sizes, int n_in,
                              void* d_out, int out_size, void* d_ws, size_t ws_size,
                              hipStream_t stream) {
    const float* feature = (const float*)d_in[0];
    const int*   eidx    = (const int*)d_in[1];   // [2, N_EDGES] int32
    // d_in[2] edge_type: unused by the reference math
    const float* W_in  = (const float*)d_in[3];
    const float* b_in  = (const float*)d_in[4];
    const float* W1_l  = (const float*)d_in[5];
    const float* b1_l  = (const float*)d_in[6];
    const float* W1_r  = (const float*)d_in[7];
    const float* W2_l  = (const float*)d_in[8];
    const float* b2_l  = (const float*)d_in[9];
    const float* W2_r  = (const float*)d_in[10];
    const float* W_out = (const float*)d_in[11];
    const float* b_out = (const float*)d_in[12];

    const int* src = eidx;
    const int* dst = eidx + N_EDGES;

    const long long NB  = (long long)N_NODES * HID;   // 25.6M elements (dense math)
    const long long NBP = (long long)M_PAD  * HID;    // padded activation rows
    char* ws = (char*)d_ws;
    auto take = [&](size_t bytes) -> char* {
        char* p = ws;
        ws += (bytes + 255) & ~(size_t)255;
        return p;
    };
    bf16*  Xa   = (bf16*)take(NBP * 2);         // activations (ping, padded)
    bf16*  Xb   = (bf16*)take(NBP * 2);         // activations (pong, padded)
    bf16*  Mb   = (bf16*)take(NBP * 2);         // bf16 mean-aggregated feats (padded)
    float* AGG  = (float*)take(NB * 4);         // f32 scatter accumulator
    float* CNT  = (float*)take((size_t)N_NODES * 4);
    bf16*  WinT = (bf16*)take((size_t)HID * HID * 2);
    bf16*  W1lT = (bf16*)take((size_t)HID * HID * 2);
    bf16*  W1rT = (bf16*)take((size_t)HID * HID * 2);
    bf16*  W2lT = (bf16*)take((size_t)HID * HID * 2);
    bf16*  W2rT = (bf16*)take((size_t)HID * HID * 2);

    const int BLK = 256;
    const int gNB   = (int)((NB + BLK - 1) / BLK);                 // 100000
    const int gW    = (HID * HID) / BLK;                           // 1024
    const int gEdge = (N_EDGES + BLK - 1) / BLK;                   // 6250
    const int gScat = (int)(((long long)N_EDGES * 128) / BLK);     // 800000
    dim3 gemmGrid(HID / 128, M_PAD / 256);                         // (4, 196)

    // weights -> transposed bf16 (once per call)
    wt_transpose<<<gW, BLK, 0, stream>>>(W_in, WinT);
    wt_transpose<<<gW, BLK, 0, stream>>>(W1_l, W1lT);
    wt_transpose<<<gW, BLK, 0, stream>>>(W1_r, W1rT);
    wt_transpose<<<gW, BLK, 0, stream>>>(W2_l, W2lT);
    wt_transpose<<<gW, BLK, 0, stream>>>(W2_r, W2rT);

    // degree counts (same for both conv layers)
    zero_f32<<<(N_NODES + BLK - 1) / BLK, BLK, 0, stream>>>(CNT, N_NODES);
    count_edges<<<gEdge, BLK, 0, stream>>>(dst, CNT);

    // x0 = leakyrelu(feature @ W_in + b_in)   -> Xb
    f32_to_bf16<<<gNB, BLK, 0, stream>>>(feature, Xa, NB);
    gemm_bf16<<<gemmGrid, BLK, 0, stream>>>(Xa, WinT, nullptr, nullptr, b_in, Xb,
                                            N_NODES, 1);

    // sage1: x1 = mean @ W1_l + b1 + x0 @ W1_r   -> Xa
    zero_f32<<<gNB, BLK, 0, stream>>>(AGG, NB);
    scatter_bf16<<<gScat, BLK, 0, stream>>>(Xb, src, dst, AGG);
    mean_bf16<<<gNB, BLK, 0, stream>>>(AGG, CNT, Mb);
    gemm_bf16<<<gemmGrid, BLK, 0, stream>>>(Mb, W1lT, Xb, W1rT, b1_l, Xa,
                                            N_NODES, 0);

    // sage2: x2 = mean @ W2_l + b2 + x1 @ W2_r   -> Xb
    zero_f32<<<gNB, BLK, 0, stream>>>(AGG, NB);
    scatter_bf16<<<gScat, BLK, 0, stream>>>(Xa, src, dst, AGG);
    mean_bf16<<<gNB, BLK, 0, stream>>>(AGG, CNT, Mb);
    gemm_bf16<<<gemmGrid, BLK, 0, stream>>>(Mb, W2lT, Xa, W2rT, b2_l, Xb,
                                            N_NODES, 0);

    // out = x2 @ W_out + b_out
    out_head<<<(N_NODES * 4 + BLK - 1) / BLK, BLK, 0, stream>>>(Xb, W_out, b_out,
                                                               (float*)d_out);
}
